// BioDecoder_18769007084051
// MI455X (gfx1250) — matrix operand order repaired and checked by reading, not compiled, not run on hardware
//
#include <hip/hip_runtime.h>
#include <hip/hip_bf16.h>
#include <math.h>

// ---------------- problem constants (from reference) ----------------
#define VSZ 32000
#define ESZ 512
#define HSZ 1024
#define DVSZ 512
#define BSZ 64
#define KSZ 49
#define MAXLEN 64
#define TEMP_INV 2.5f   // 1/0.4
#define TOPP 0.7f

typedef __bf16 bf16_t;
typedef bf16_t bf16x16 __attribute__((ext_vector_type(16)));
typedef bf16_t bf16x8  __attribute__((ext_vector_type(8)));
typedef float  f32x8   __attribute__((ext_vector_type(8)));

__device__ __forceinline__ unsigned short f2bf(float f) {
    unsigned u = __float_as_uint(f);
    unsigned r = u + 0x7fffu + ((u >> 16) & 1u);   // round-to-nearest-even
    return (unsigned short)(r >> 16);
}

__device__ __forceinline__ float sigm(float x) { return 1.0f / (1.0f + __expf(-x)); }

__device__ __forceinline__ float u01(unsigned long long x) {
    x += 0x9E3779B97F4A7C15ULL;
    x = (x ^ (x >> 30)) * 0xBF58476D1CE4E5B9ULL;
    x = (x ^ (x >> 27)) * 0x94D049BB133111EBULL;
    x ^= x >> 31;
    return (float)(x >> 40) * (1.0f / 16777216.0f);
}

// ---------------- WMMA fragment loaders (CDNA5 wave32 layouts, ISA 7.12.2) --
// A (16x32 bf16): lane L holds row M=L%16; elems 0..7 -> K=kb+(L>=16?8:0)+e,
//                 elems 8..15 -> K=kb+16+(L>=16?8:0)+(e-8)
__device__ __forceinline__ bf16x16 load_a_frag(const bf16_t* A, int lda,
                                               int mbase, int kb, int lane) {
    int m = mbase + (lane & 15);
    int koff = kb + ((lane >> 4) << 3);
    const bf16x8* p0 = (const bf16x8*)(A + (size_t)m * lda + koff);
    const bf16x8* p1 = (const bf16x8*)(A + (size_t)m * lda + koff + 16);
    bf16x8 lo = *p0, hi = *p1;
    bf16x16 r;
#pragma unroll
    for (int e = 0; e < 8; ++e) { r[e] = lo[e]; r[e + 8] = hi[e]; }
    return r;
}
// B (32x16 bf16 = W^T, W[n,k] row-major): lane L holds col N=L%16;
// elems 0..7 -> K=kb+(L>=16?8:0)+e, elems 8..15 -> K=kb+16+(L>=16?8:0)+(e-8)
__device__ __forceinline__ bf16x16 load_b_frag(const bf16_t* W, int ldw,
                                               int nbase, int kb, int lane) {
    int n = nbase + (lane & 15);
    int koff = kb + ((lane >> 4) << 3);
    const bf16x8* p0 = (const bf16x8*)(W + (size_t)n * ldw + koff);
    bf16x8 lo = p0[0], hi = p0[2];
    bf16x16 r;
#pragma unroll
    for (int e = 0; e < 8; ++e) { r[e] = lo[e]; r[e + 8] = hi[e]; }
    return r;
}

// ---------------- async staging of a contiguous panel into LDS --------------
// CDNA5 GLOBAL_LOAD_ASYNC_TO_LDS_B128: per-lane 16B memory->LDS, ASYNCcnt.
__device__ __forceinline__ void stage_async(const bf16_t* g, unsigned lds_off_base,
                                            int bytes, int tid, int nthr) {
    const char* gp = (const char*)g;
    for (int off = tid * 16; off < bytes; off += nthr * 16) {
        unsigned lds_off = lds_off_base + (unsigned)off;
        unsigned long long ga = (unsigned long long)(const void*)(gp + off);
        asm volatile("global_load_async_to_lds_b128 %0, %1, off"
                     :: "v"(lds_off), "v"(ga) : "memory");
    }
}

// ---------------- bf16 WMMA GEMM: C = A*WA^T (+ B2*WB^T) + bias -------------
// M mult of 64, N mult of 64, Ka/Kb mult of 32. Block = 128 thr (4 waves).
// Block covers 64 rows (whole M-block) x 64 cols; each wave: 4 M-tiles x 1 N-tile,
// so each streamed B (weight) fragment is reused 4x. A panels live in LDS,
// staged via async-to-LDS; B is software-pipelined from global.
extern __shared__ char smem_raw[];

__device__ __forceinline__ void gemm_panel(const bf16_t* As, const bf16_t* W, int K,
                                           int nbase, int lane, f32x8* acc) {
    bf16x16 bcur = load_b_frag(W, K, nbase, 0, lane);
    for (int k = 32; k < K; k += 32) {
        bf16x16 bnxt = load_b_frag(W, K, nbase, k, lane);
        __builtin_prefetch(W + (size_t)(nbase + (lane & 15)) * K + k + 64, 0, 3);
        bf16x16 a0 = load_a_frag(As, K, 0,  k - 32, lane);
        bf16x16 a1 = load_a_frag(As, K, 16, k - 32, lane);
        bf16x16 a2 = load_a_frag(As, K, 32, k - 32, lane);
        bf16x16 a3 = load_a_frag(As, K, 48, k - 32, lane);
        acc[0] = __builtin_amdgcn_wmma_f32_16x16x32_bf16(false, a0, false, bcur, (short)0, acc[0], false, false);
        acc[1] = __builtin_amdgcn_wmma_f32_16x16x32_bf16(false, a1, false, bcur, (short)0, acc[1], false, false);
        acc[2] = __builtin_amdgcn_wmma_f32_16x16x32_bf16(false, a2, false, bcur, (short)0, acc[2], false, false);
        acc[3] = __builtin_amdgcn_wmma_f32_16x16x32_bf16(false, a3, false, bcur, (short)0, acc[3], false, false);
        bcur = bnxt;
    }
    bf16x16 a0 = load_a_frag(As, K, 0,  K - 32, lane);
    bf16x16 a1 = load_a_frag(As, K, 16, K - 32, lane);
    bf16x16 a2 = load_a_frag(As, K, 32, K - 32, lane);
    bf16x16 a3 = load_a_frag(As, K, 48, K - 32, lane);
    acc[0] = __builtin_amdgcn_wmma_f32_16x16x32_bf16(false, a0, false, bcur, (short)0, acc[0], false, false);
    acc[1] = __builtin_amdgcn_wmma_f32_16x16x32_bf16(false, a1, false, bcur, (short)0, acc[1], false, false);
    acc[2] = __builtin_amdgcn_wmma_f32_16x16x32_bf16(false, a2, false, bcur, (short)0, acc[2], false, false);
    acc[3] = __builtin_amdgcn_wmma_f32_16x16x32_bf16(false, a3, false, bcur, (short)0, acc[3], false, false);
}

__global__ void wmma_gemm_bias(const bf16_t* __restrict__ A,
                               const bf16_t* __restrict__ WA, int Ka,
                               const bf16_t* __restrict__ B2,
                               const bf16_t* __restrict__ WB, int Kb,
                               const float* __restrict__ bias,
                               float* __restrict__ C,
                               unsigned short* __restrict__ Cbf,
                               int M, int N) {
    const int tid = threadIdx.x;
    const int lane = tid & 31;
    const int wave = tid >> 5;
    const int nbase = (blockIdx.x * 4 + wave) * 16;
    const int mblock = blockIdx.y * 64;

    const unsigned lds_base = (unsigned)(unsigned long long)(void*)smem_raw;
    const int bytesA = 64 * Ka * 2;
    stage_async(A + (size_t)mblock * Ka, lds_base, bytesA, tid, blockDim.x);
    if (B2 != nullptr)
        stage_async(B2 + (size_t)mblock * Kb, lds_base + (unsigned)bytesA,
                    64 * Kb * 2, tid, blockDim.x);
    asm volatile("s_wait_asynccnt 0x0" ::: "memory");
    __syncthreads();

    const bf16_t* ApanelA = (const bf16_t*)smem_raw;
    const bf16_t* ApanelB = (const bf16_t*)(smem_raw + bytesA);

    f32x8 acc[4] = {};
    gemm_panel(ApanelA, WA, Ka, nbase, lane, acc);
    if (B2 != nullptr)
        gemm_panel(ApanelB, WB, Kb, nbase, lane, acc);

    const int n = nbase + (lane & 15);
    const float bv = bias ? bias[n] : 0.0f;
#pragma unroll
    for (int mt = 0; mt < 4; ++mt) {
        const int mrow0 = mblock + mt * 16 + ((lane >> 4) << 3);
#pragma unroll
        for (int r = 0; r < 8; ++r) {
            float v = acc[mt][r] + bv;
            size_t idx = (size_t)(mrow0 + r) * N + n;
            C[idx] = v;
            if (Cbf) Cbf[idx] = f2bf(v);
        }
    }
}

// ---------------- small kernels ---------------------------------------------
__global__ void f32_to_bf16_kernel(const float* __restrict__ s,
                                   unsigned short* __restrict__ d, long long n) {
    long long i = (long long)blockIdx.x * blockDim.x + threadIdx.x;
    if (i < n) d[i] = f2bf(s[i]);
}

__global__ void addvec_kernel(const float* a, const float* b, float* o, int n) {
    int i = blockIdx.x * blockDim.x + threadIdx.x;
    if (i < n) o[i] = a[i] + b[i];
}

__global__ void init_state_kernel(const float* __restrict__ thought,
                                  float* h1, float* c1, float* h2, float* c2,
                                  unsigned short* h1b, unsigned short* h2b) {
    int u = blockIdx.x * blockDim.x + threadIdx.x;
    int b = blockIdx.y;
    float v = thought[(size_t)b * HSZ + u];
    size_t i = (size_t)b * HSZ + u;
    h1[i] = v; h2[i] = v; c1[i] = 0.f; c2[i] = 0.f;
    unsigned short bv = f2bf(v);
    h1b[i] = bv; h2b[i] = bv;
}

__global__ void init_tok_kernel(int* next_tok, int* hist, int* histv) {
    int b = threadIdx.x;
    if (b < BSZ) {
        next_tok[b] = 1;
        for (int j = 0; j < 3; ++j) { hist[b * 3 + j] = 0; histv[b * 3 + j] = 0; }
    }
}

__global__ void emb_gather_kernel(const float* __restrict__ emb,
                                  const int* __restrict__ tok,
                                  unsigned short* __restrict__ out) {
    int e = blockIdx.x * blockDim.x + threadIdx.x;
    int b = blockIdx.y;
    if (e < ESZ)
        out[(size_t)b * ESZ + e] = f2bf(emb[(size_t)tok[b] * ESZ + e]);
}

// z gate order: i, f, g, o
__global__ void lstm_elem_kernel(const float* __restrict__ z,
                                 float* __restrict__ h, float* __restrict__ c,
                                 unsigned short* __restrict__ hb) {
    int u = blockIdx.x * blockDim.x + threadIdx.x;
    int b = blockIdx.y;
    size_t zb = (size_t)b * 4 * HSZ;
    float zi = z[zb + u], zf = z[zb + HSZ + u];
    float zg = z[zb + 2 * HSZ + u], zo = z[zb + 3 * HSZ + u];
    size_t i = (size_t)b * HSZ + u;
    float cn = sigm(zf) * c[i] + sigm(zi) * tanhf(zg);
    c[i] = cn;
    float hn = sigm(zo) * tanhf(cn);
    h[i] = hn;
    hb[i] = f2bf(hn);
}

__global__ void layernorm_kernel(float* __restrict__ x,
                                 const float* __restrict__ g,
                                 const float* __restrict__ bb) {
    __shared__ float r1[256], r2[256];
    int row = blockIdx.x, tid = threadIdx.x;
    float* p = x + (size_t)row * HSZ;
    float s = 0.f, sq = 0.f;
    for (int i = tid; i < HSZ; i += 256) { float v = p[i]; s += v; sq += v * v; }
    r1[tid] = s; r2[tid] = sq; __syncthreads();
    for (int st = 128; st > 0; st >>= 1) {
        if (tid < st) { r1[tid] += r1[tid + st]; r2[tid] += r2[tid + st]; }
        __syncthreads();
    }
    float mean = r1[0] * (1.0f / HSZ);
    float var = r2[0] * (1.0f / HSZ) - mean * mean;
    float inv = rsqrtf(var + 1e-5f);
    for (int i = tid; i < HSZ; i += 256)
        p[i] = (p[i] - mean) * inv * g[i] + bb[i];
}

// gate = sigmoid(tanh(g1) . Wg2 + bg2); out = h2*gate + attn(vp)
__global__ void gate_attn_kernel(const float* __restrict__ h2,
                                 const float* __restrict__ g1,
                                 const float* __restrict__ Wg2,
                                 const float* __restrict__ bg2,
                                 const float* __restrict__ vp,
                                 float* __restrict__ outf,
                                 unsigned short* __restrict__ outb) {
    __shared__ float red[256];
    __shared__ float satt[64];
    __shared__ float sgate;
    int b = blockIdx.x, tid = threadIdx.x;
    float acc = 0.f;
    for (int u = tid; u < HSZ; u += 256)
        acc += tanhf(g1[(size_t)b * HSZ + u]) * Wg2[u];
    red[tid] = acc; __syncthreads();
    for (int s = 128; s > 0; s >>= 1) {
        if (tid < s) red[tid] += red[tid + s];
        __syncthreads();
    }
    if (tid == 0) sgate = sigm(red[0] + bg2[0]);
    __syncthreads();
    float gate = sgate;
    const float scale = rsqrtf((float)HSZ);
    for (int k = 0; k < KSZ; ++k) {
        const float* vpr = vp + ((size_t)b * KSZ + k) * HSZ;
        float a = 0.f;
        for (int u = tid; u < HSZ; u += 256)
            a += gate * h2[(size_t)b * HSZ + u] * vpr[u];
        red[tid] = a; __syncthreads();
        for (int s = 128; s > 0; s >>= 1) {
            if (tid < s) red[tid] += red[tid + s];
            __syncthreads();
        }
        if (tid == 0) satt[k] = red[0] * scale;
        __syncthreads();
    }
    if (tid == 0) {
        float m = -3.4e38f;
        for (int k = 0; k < KSZ; ++k) m = fmaxf(m, satt[k]);
        float z = 0.f;
        for (int k = 0; k < KSZ; ++k) { satt[k] = __expf(satt[k] - m); z += satt[k]; }
        float iz = 1.0f / z;
        for (int k = 0; k < KSZ; ++k) satt[k] *= iz;
    }
    __syncthreads();
    for (int u = tid; u < HSZ; u += 256) {
        float o = gate * h2[(size_t)b * HSZ + u];
        float av = 0.f;
        for (int k = 0; k < KSZ; ++k)
            av += satt[k] * vp[((size_t)b * KSZ + k) * HSZ + u];
        float v = o + av;
        outf[(size_t)b * HSZ + u] = v;
        outb[(size_t)b * HSZ + u] = f2bf(v);
    }
}

// penalty + temperature + nucleus-threshold search + inverse-CDF sampling
__global__ __launch_bounds__(1024)
void sample_kernel(float* __restrict__ logits, int* __restrict__ hist,
                   int* __restrict__ histv, int* __restrict__ next_tok,
                   int* __restrict__ out_toks, int t, unsigned long long seed) {
    const int b = blockIdx.x, tid = threadIdx.x;
    const int NT = 1024;
    float* L = logits + (size_t)b * VSZ;
    __shared__ float sred[1024];
    __shared__ int sidx[1024];
    __shared__ float s_max, s_Z;
    __shared__ int s_arg;
    __shared__ float s_run;
    __shared__ int s_chosen;

    if (tid == 0) {
        for (int j = 0; j < 3; ++j) {
            if (histv[b * 3 + j]) {
                int tok = hist[b * 3 + j];
                bool dup = false;
                for (int jj = 0; jj < j; ++jj)
                    if (histv[b * 3 + jj] && hist[b * 3 + jj] == tok) dup = true;
                if (!dup) L[tok] -= 2.0f;
            }
        }
    }
    __syncthreads();
    float m = -3.4e38f; int mi = 0;
    for (int i = tid; i < VSZ; i += NT) { float v = L[i]; if (v > m) { m = v; mi = i; } }
    sred[tid] = m; sidx[tid] = mi; __syncthreads();
    for (int s = NT >> 1; s > 0; s >>= 1) {
        if (tid < s && sred[tid + s] > sred[tid]) { sred[tid] = sred[tid + s]; sidx[tid] = sidx[tid + s]; }
        __syncthreads();
    }
    if (tid == 0) { s_max = sred[0]; s_arg = sidx[0]; }
    __syncthreads();
    m = s_max;
    float z = 0.f;
    for (int i = tid; i < VSZ; i += NT) z += __expf((L[i] - m) * TEMP_INV);
    sred[tid] = z; __syncthreads();
    for (int s = NT >> 1; s > 0; s >>= 1) {
        if (tid < s) sred[tid] += sred[tid + s];
        __syncthreads();
    }
    if (tid == 0) s_Z = sred[0];
    __syncthreads();
    float iZ = 1.0f / s_Z;
    for (int i = tid; i < VSZ; i += NT) L[i] = __expf((L[i] - m) * TEMP_INV) * iZ;
    __syncthreads();
    float lo = 0.f, hi = 1.f;
    for (int it = 0; it < 24; ++it) {
        float mid = 0.5f * (lo + hi);
        float sm = 0.f;
        for (int i = tid; i < VSZ; i += NT) { float p = L[i]; sm += (p >= mid) ? p : 0.f; }
        sred[tid] = sm; __syncthreads();
        for (int s = NT >> 1; s > 0; s >>= 1) {
            if (tid < s) sred[tid] += sred[tid + s];
            __syncthreads();
        }
        float S = sred[0]; __syncthreads();
        if (S >= TOPP) lo = mid; else hi = mid;
    }
    float tau = lo;
    float sm = 0.f;
    for (int i = tid; i < VSZ; i += NT) { float p = L[i]; sm += (p >= tau) ? p : 0.f; }
    sred[tid] = sm; __syncthreads();
    for (int s = NT >> 1; s > 0; s >>= 1) {
        if (tid < s) sred[tid] += sred[tid + s];
        __syncthreads();
    }
    float kept = sred[0]; __syncthreads();
    float u = u01(seed ^ ((unsigned long long)t * 9973ULL + (unsigned long long)b * 7919ULL)) * kept;
    if (tid == 0) { s_run = 0.f; s_chosen = 0x7fffffff; }
    __syncthreads();
    for (int base = 0; base < VSZ; base += NT) {
        int i = base + tid;
        float p = (i < VSZ) ? L[i] : 0.f;
        if (p < tau) p = 0.f;
        sred[tid] = p; __syncthreads();
        for (int off = 1; off < NT; off <<= 1) {
            float v = (tid >= off) ? sred[tid - off] : 0.f;
            __syncthreads();
            sred[tid] += v;
            __syncthreads();
        }
        float incl = s_run + sred[tid];
        float excl = incl - p;
        if (p > 0.f && excl < u && u <= incl) atomicMin(&s_chosen, i);
        __syncthreads();
        if (tid == 0) s_run += sred[NT - 1];
        __syncthreads();
        if (s_chosen != 0x7fffffff) break;
    }
    int tok = (s_chosen != 0x7fffffff) ? s_chosen : s_arg;
    if (tid == 0) {
        next_tok[b] = tok;
        out_toks[(size_t)b * MAXLEN + t] = tok;
        hist[b * 3] = hist[b * 3 + 1]; hist[b * 3 + 1] = hist[b * 3 + 2]; hist[b * 3 + 2] = tok;
        histv[b * 3] = histv[b * 3 + 1]; histv[b * 3 + 1] = histv[b * 3 + 2]; histv[b * 3 + 2] = 1;
    }
}

// ---------------- host orchestration ----------------------------------------
static void launch_convert(const float* s, unsigned short* d, long long n, hipStream_t st) {
    f32_to_bf16_kernel<<<dim3((unsigned)((n + 255) / 256)), dim3(256), 0, st>>>(s, d, n);
}

static void launch_gemm(const unsigned short* A, const unsigned short* WA, int Ka,
                        const unsigned short* B2, const unsigned short* WB, int Kb,
                        const float* bias, float* C, unsigned short* Cbf,
                        int M, int N, hipStream_t st) {
    dim3 g(N / 64, M / 64), blk(128);
    size_t smem = (size_t)64 * Ka * 2 + (B2 ? (size_t)64 * Kb * 2 : 0);
    wmma_gemm_bias<<<g, blk, smem, st>>>((const bf16_t*)A, (const bf16_t*)WA, Ka,
                                         (const bf16_t*)B2, (const bf16_t*)WB, Kb,
                                         bias, C, Cbf, M, N);
}

extern "C" void kernel_launch(void* const* d_in, const int* in_sizes, int n_in,
                              void* d_out, int out_size, void* d_ws, size_t ws_size,
                              hipStream_t stream) {
    const float* thought  = (const float*)d_in[0];
    const float* visual   = (const float*)d_in[1];
    /* d_in[2] = max_len scalar (known: 64) */
    const float* embedding = (const float*)d_in[3];
    const float* Wih1 = (const float*)d_in[4];
    const float* Whh1 = (const float*)d_in[5];
    const float* bih1 = (const float*)d_in[6];
    const float* bhh1 = (const float*)d_in[7];
    const float* Wih2 = (const float*)d_in[8];
    const float* Whh2 = (const float*)d_in[9];
    const float* bih2 = (const float*)d_in[10];
    const float* bhh2 = (const float*)d_in[11];
    const float* Wg1 = (const float*)d_in[12];
    const float* bg1 = (const float*)d_in[13];
    const float* Wg2 = (const float*)d_in[14];
    const float* bg2 = (const float*)d_in[15];
    const float* Wo = (const float*)d_in[16];
    const float* bo = (const float*)d_in[17];
    const float* Wv = (const float*)d_in[18];
    const float* bv = (const float*)d_in[19];
    const float* ln_g = (const float*)d_in[20];
    const float* ln_b = (const float*)d_in[21];
    int* out_toks = (int*)d_out;   // [B, MAXLEN] int32 tokens

    char* p = (char*)d_ws;
    auto alloc = [&](size_t bytes) -> void* {
        void* r = (void*)p;
        p += (bytes + 255) & ~(size_t)255;
        return r;
    };
    unsigned short* Wih1b = (unsigned short*)alloc((size_t)4 * HSZ * ESZ * 2);
    unsigned short* Whh1b = (unsigned short*)alloc((size_t)4 * HSZ * HSZ * 2);
    unsigned short* Wih2b = (unsigned short*)alloc((size_t)4 * HSZ * HSZ * 2);
    unsigned short* Whh2b = (unsigned short*)alloc((size_t)4 * HSZ * HSZ * 2);
    unsigned short* Wg1b  = (unsigned short*)alloc((size_t)HSZ * HSZ * 2);
    unsigned short* Wob   = (unsigned short*)alloc((size_t)VSZ * HSZ * 2);
    unsigned short* Wvb   = (unsigned short*)alloc((size_t)HSZ * DVSZ * 2);
    unsigned short* visb  = (unsigned short*)alloc((size_t)BSZ * KSZ * DVSZ * 2);
    float* vp    = (float*)alloc((size_t)BSZ * KSZ * HSZ * 4);
    float* zbuf  = (float*)alloc((size_t)BSZ * 4 * HSZ * 4);
    float* h1    = (float*)alloc((size_t)BSZ * HSZ * 4);
    float* c1    = (float*)alloc((size_t)BSZ * HSZ * 4);
    float* h2    = (float*)alloc((size_t)BSZ * HSZ * 4);
    float* c2    = (float*)alloc((size_t)BSZ * HSZ * 4);
    float* g1    = (float*)alloc((size_t)BSZ * HSZ * 4);
    float* outf  = (float*)alloc((size_t)BSZ * HSZ * 4);
    float* logits = (float*)alloc((size_t)BSZ * VSZ * 4);
    float* bsum1 = (float*)alloc((size_t)4 * HSZ * 4);
    float* bsum2 = (float*)alloc((size_t)4 * HSZ * 4);
    unsigned short* h1b  = (unsigned short*)alloc((size_t)BSZ * HSZ * 2);
    unsigned short* h2b  = (unsigned short*)alloc((size_t)BSZ * HSZ * 2);
    unsigned short* outb = (unsigned short*)alloc((size_t)BSZ * HSZ * 2);
    unsigned short* embb = (unsigned short*)alloc((size_t)BSZ * ESZ * 2);
    int* hist     = (int*)alloc((size_t)BSZ * 3 * 4);
    int* histv    = (int*)alloc((size_t)BSZ * 3 * 4);
    int* next_tok = (int*)alloc((size_t)BSZ * 4);
    (void)ws_size; (void)n_in; (void)in_sizes; (void)out_size;

    launch_convert(Wih1, Wih1b, (long long)4 * HSZ * ESZ, stream);
    launch_convert(Whh1, Whh1b, (long long)4 * HSZ * HSZ, stream);
    launch_convert(Wih2, Wih2b, (long long)4 * HSZ * HSZ, stream);
    launch_convert(Whh2, Whh2b, (long long)4 * HSZ * HSZ, stream);
    launch_convert(Wg1, Wg1b, (long long)HSZ * HSZ, stream);
    launch_convert(Wo, Wob, (long long)VSZ * HSZ, stream);
    launch_convert(Wv, Wvb, (long long)HSZ * DVSZ, stream);
    launch_convert(visual, visb, (long long)BSZ * KSZ * DVSZ, stream);
    addvec_kernel<<<dim3(16), dim3(256), 0, stream>>>(bih1, bhh1, bsum1, 4 * HSZ);
    addvec_kernel<<<dim3(16), dim3(256), 0, stream>>>(bih2, bhh2, bsum2, 4 * HSZ);
    init_state_kernel<<<dim3(HSZ / 256, BSZ), dim3(256), 0, stream>>>(
        thought, h1, c1, h2, c2, h1b, h2b);
    init_tok_kernel<<<dim3(1), dim3(64), 0, stream>>>(next_tok, hist, histv);

    // visual projection (WMMA) + LayerNorm
    launch_gemm(visb, Wvb, DVSZ, nullptr, nullptr, 0, bv, vp, nullptr,
                BSZ * KSZ, HSZ, stream);
    layernorm_kernel<<<dim3(BSZ * KSZ), dim3(256), 0, stream>>>(vp, ln_g, ln_b);

    for (int t = 0; t < MAXLEN; ++t) {
        emb_gather_kernel<<<dim3(ESZ / 256, BSZ), dim3(256), 0, stream>>>(
            embedding, next_tok, embb);
        launch_gemm(embb, Wih1b, ESZ, h1b, Whh1b, HSZ, bsum1, zbuf, nullptr,
                    BSZ, 4 * HSZ, stream);
        lstm_elem_kernel<<<dim3(HSZ / 256, BSZ), dim3(256), 0, stream>>>(zbuf, h1, c1, h1b);
        launch_gemm(h1b, Wih2b, HSZ, h2b, Whh2b, HSZ, bsum2, zbuf, nullptr,
                    BSZ, 4 * HSZ, stream);
        lstm_elem_kernel<<<dim3(HSZ / 256, BSZ), dim3(256), 0, stream>>>(zbuf, h2, c2, h2b);
        launch_gemm(h2b, Wg1b, HSZ, nullptr, nullptr, 0, bg1, g1, nullptr,
                    BSZ, HSZ, stream);
        gate_attn_kernel<<<dim3(BSZ), dim3(256), 0, stream>>>(
            h2, g1, Wg2, bg2, vp, outf, outb);
        launch_gemm(outb, Wob, HSZ, nullptr, nullptr, 0, bo, logits, nullptr,
                    BSZ, VSZ, stream);
        sample_kernel<<<dim3(BSZ), dim3(1024), 0, stream>>>(
            logits, hist, histv, next_tok, out_toks, t, 0x5DEECE66DULL);
    }
}
